// ESNPCA_69578470195873
// MI455X (gfx1250) — matrix operand order repaired and checked by reading, not compiled, not run on hardware
//
#include <hip/hip_runtime.h>
#include <cstdint>
#include <cstddef>

#define R_SIZE   2048
#define I_SIZE   64
#define EXT      2112          // R + I
#define KPAD     2304          // EXT padded to 72 K-chunks of 32 (zero-filled pad)
#define T_STEPS  8192
#define WARM     100
#define LRATE    0.01f
#define NWG      128           // persistent workgroups; 16 rows each
#define ROWS_WG  16
#define THREADS  256           // 8 waves (wave32)
#define WAVES    8
#define CHUNKS_PER_WAVE 9      // KPAD/32/WAVES — uniform trip count, full unroll
#define SW_STRIDE 2312         // 2304 + 8 halves pad (minimal LDS bank cycles)

// workspace layout (bytes)
#define CTR_OFF   0
#define HF32_OFF  256
#define HB_OFF    8448         // two bf16 h buffers (double-buffered), 4096 B each
#define WEXT_OFF  24576        // 2048 x 2112 bf16 = 8,650,752 B

typedef __attribute__((ext_vector_type(16))) __bf16 bf16x16;
typedef __attribute__((ext_vector_type(8)))  __bf16 bf16x8;
typedef __attribute__((ext_vector_type(8)))  float  f32x8;

// ---- init: zero barrier counter + h state (runs every launch -> graph-replay safe)
__global__ void esn_init(float* __restrict__ hf32, __bf16* __restrict__ hb,
                         unsigned int* __restrict__ ctr) {
    int tid = threadIdx.x;
    if (tid == 0) *ctr = 0u;
    for (int i = tid; i < R_SIZE; i += blockDim.x) {
        hf32[i] = 0.0f;
        hb[i]           = (__bf16)0.0f;   // buffer 0
        hb[R_SIZE + i]  = (__bf16)0.0f;   // buffer 1
    }
}

// ---- pack [Wh | Win] rows into one row-major bf16 matrix 2048 x 2112
__global__ void esn_pack(const float* __restrict__ Win, const float* __restrict__ Wh,
                         __bf16* __restrict__ Wext) {
    int row = blockIdx.x;
    for (int c = threadIdx.x; c < EXT; c += blockDim.x) {
        float v = (c < R_SIZE) ? Wh[(size_t)row * R_SIZE + c]
                               : Win[(size_t)row * I_SIZE + (c - R_SIZE)];
        Wext[(size_t)row * EXT + c] = (__bf16)v;
    }
}

// ---- persistent recurrence kernel
__global__ void __launch_bounds__(THREADS, 1)
esn_run(const float* __restrict__ x, const __bf16* __restrict__ Wext,
        float* __restrict__ hf32, __bf16* __restrict__ hb,
        unsigned int* __restrict__ ctr, float* __restrict__ out) {
    __shared__ __align__(16) __bf16 sW[ROWS_WG * SW_STRIDE];   // 73,984 B weight slice
    __shared__ __align__(32) __bf16 sHx[KPAD];                 // [h ; x_t ; 0-pad] bf16
    __shared__ float sRed[WAVES * ROWS_WG];                    // cross-wave reduction

    const int g    = blockIdx.x;
    const int tid  = threadIdx.x;
    const int wave = tid >> 5;
    const int lane = tid & 31;
    const int m    = lane & 15;          // A-matrix row owned by this lane
    const int hi   = lane >> 4;          // lane-half selector (K sub-chunk)

    // stage 16 weight rows into LDS (vectorized b128); zero the K-pad region
    for (int idx = tid; idx < ROWS_WG * (KPAD / 8); idx += THREADS) {
        int r  = idx / (KPAD / 8);
        int c8 = idx - r * (KPAD / 8);
        uint4 v = make_uint4(0u, 0u, 0u, 0u);
        if (c8 < EXT / 8)
            v = *(const uint4*)(Wext + ((size_t)(g * ROWS_WG + r) * EXT + c8 * 8));
        *(uint4*)(&sW[r * SW_STRIDE + c8 * 8]) = v;
    }
    // zero the K-pad of the state vector once (never touched again)
    if (tid < KPAD - EXT) sHx[EXT + tid] = (__bf16)0.0f;
    __syncthreads();

    for (int t = 0; t < T_STEPS; ++t) {
        const __bf16* hin  = hb + (size_t)(t & 1) * R_SIZE;
        __bf16*       hout = hb + (size_t)((t + 1) & 1) * R_SIZE;

        // gather extended state [h ; x_t] into LDS
        ((uint4*)sHx)[tid] = ((const uint4*)hin)[tid];          // 2048 bf16
        if (tid < I_SIZE) sHx[R_SIZE + tid] = (__bf16)x[(size_t)t * I_SIZE + tid];
        if (tid == 0 && t + 1 < T_STEPS)
            __builtin_prefetch(&x[(size_t)(t + 1) * I_SIZE], 0, 0);
        __syncthreads();

        // WMMA matvec: 9 chunks per wave, uniform trip count, two accumulators for ILP
        f32x8 acc0 = {0.f, 0.f, 0.f, 0.f, 0.f, 0.f, 0.f, 0.f};
        f32x8 acc1 = {0.f, 0.f, 0.f, 0.f, 0.f, 0.f, 0.f, 0.f};
#pragma unroll
        for (int j = 0; j < CHUNKS_PER_WAVE; ++j) {
            const int kc = wave + j * WAVES;
            const __bf16* ap = &sW[m * SW_STRIDE + kc * 32 + hi * 8];
            const __bf16* bp = &sHx[kc * 32 + hi * 8];
            union { bf16x8 h[2]; bf16x16 v; } ua, ub;
            ua.h[0] = *(const bf16x8*)(ap);        // K = base .. base+7
            ua.h[1] = *(const bf16x8*)(ap + 16);   // K = base+16 .. base+23
            ub.h[0] = *(const bf16x8*)(bp);        // h chunk broadcast to all N columns
            ub.h[1] = *(const bf16x8*)(bp + 16);
            if (j & 1)
                acc1 = __builtin_amdgcn_wmma_f32_16x16x32_bf16(
                    false, ua.v, false, ub.v, (short)0, acc1, false, false);
            else
                acc0 = __builtin_amdgcn_wmma_f32_16x16x32_bf16(
                    false, ua.v, false, ub.v, (short)0, acc0, false, false);
        }

        // column 0 of D: lane 0 holds rows 0..7, lane 16 holds rows 8..15
        if (lane == 0) {
#pragma unroll
            for (int r = 0; r < 8; ++r) sRed[wave * 16 + r] = acc0[r] + acc1[r];
        }
        if (lane == 16) {
#pragma unroll
            for (int r = 0; r < 8; ++r) sRed[wave * 16 + 8 + r] = acc0[r] + acc1[r];
        }
        __syncthreads();

        // reduce across waves, apply tanh + leaky update, emit state
        if (tid < ROWS_WG) {
            float s = 0.f;
#pragma unroll
            for (int w = 0; w < WAVES; ++w) s += sRed[w * 16 + tid];
            int gi = g * ROWS_WG + tid;
            float hn = (1.0f - LRATE) * hf32[gi] + LRATE * tanhf(s);
            hf32[gi] = hn;
            hout[gi] = (__bf16)hn;
            if (t >= WARM) out[(size_t)(t - WARM) * EXT + gi] = hn;
        }
        if (g == 0 && t >= WARM && tid >= 64 && tid < 128) {
            int c = tid - 64;
            out[(size_t)(t - WARM) * EXT + R_SIZE + c] = x[(size_t)t * I_SIZE + c];
        }

        // release writes, grid-wide barrier, acquire for next step's h read
        __threadfence();
        __syncthreads();
        if (tid == 0) {
            __hip_atomic_fetch_add(ctr, 1u, __ATOMIC_RELEASE, __HIP_MEMORY_SCOPE_AGENT);
            unsigned int target = (unsigned int)NWG * (unsigned int)(t + 1);
            while (__hip_atomic_load(ctr, __ATOMIC_ACQUIRE, __HIP_MEMORY_SCOPE_AGENT) < target)
                __builtin_amdgcn_s_sleep(1);
        }
        __syncthreads();
        __threadfence();
    }
}

extern "C" void kernel_launch(void* const* d_in, const int* in_sizes, int n_in,
                              void* d_out, int out_size, void* d_ws, size_t ws_size,
                              hipStream_t stream) {
    const float* x   = (const float*)d_in[0];   // [T, 64]
    const float* Win = (const float*)d_in[1];   // [2048, 64]
    const float* Wh  = (const float*)d_in[2];   // [2048, 2048]
    float* out = (float*)d_out;                 // [8092, 2112]

    char* ws = (char*)d_ws;
    unsigned int* ctr  = (unsigned int*)(ws + CTR_OFF);
    float*        hf32 = (float*)(ws + HF32_OFF);
    __bf16*       hb   = (__bf16*)(ws + HB_OFF);
    __bf16*       Wext = (__bf16*)(ws + WEXT_OFF);

    esn_init<<<1, 256, 0, stream>>>(hf32, hb, ctr);
    esn_pack<<<R_SIZE, 256, 0, stream>>>(Win, Wh, Wext);
    esn_run<<<NWG, THREADS, 0, stream>>>(x, Wext, hf32, hb, ctr, out);
}
